// MLPVAD_85899345920305
// MI455X (gfx1250) — matrix operand order repaired and checked by reading, not compiled, not run on hardware
//
#include <hip/hip_runtime.h>
#include <hip/hip_bf16.h>
#include <cstdint>

// ---------------- problem constants ----------------
#define T_LEN   8192
#define BATCH   32
#define NMELS   40
#define KT1     14          // ceil(440/32)
#define NT1     8           // 128/16
#define KT2     4
#define NT2     8
#define KT3     4
#define NT3     4

#define W1_HALVES (KT1*NT1*512)   // 57344
#define W2_HALVES (KT2*NT2*512)   // 16384
#define W3_HALVES (KT3*NT3*512)   // 8192
#define W_LDS_HALVES (W1_HALVES+W2_HALVES+W3_HALVES)   // 81920 halves = 163840 B
#define STAGE_HALVES 1088          // 26*40=1040 used (+pad), 16B aligned stride
#define HBUF_HALVES  2048          // 16 x 128
#define WAVES        8
#define SMEM_HALVES  (W_LDS_HALVES + WAVES*STAGE_HALVES + WAVES*HBUF_HALVES)
#define SMEM_BYTES   (SMEM_HALVES*2)   // 214016 B < 320 KB

typedef _Float16 v16h  __attribute__((ext_vector_type(16)));
typedef _Float16 v8h   __attribute__((ext_vector_type(8)));
typedef float    v8f   __attribute__((ext_vector_type(8)));
typedef unsigned u32x4 __attribute__((ext_vector_type(4)));
typedef int      i32x8 __attribute__((ext_vector_type(8)));
typedef int      i32x4 __attribute__((ext_vector_type(4)));

#if defined(__HIP_DEVICE_COMPILE__) && __has_builtin(__builtin_amdgcn_tensor_load_to_lds) && __has_builtin(__builtin_amdgcn_s_wait_tensorcnt)
#define USE_TDM 1
#else
#define USE_TDM 0
#endif

// ---- weight pre-swizzle: row-major fp32 W[k][n] -> f16 B-operand tiles ----
// tile (kt,nt): 32 lanes x 16 halves; lane = n_local + 16*khi,
// half h<8 -> k_local = 8*khi + h, h>=8 -> k_local = 16 + 8*khi + (h-8).
__global__ void MLPVAD_prep_wtiles(const float* __restrict__ W, int Kdim, int Ndim,
                                   _Float16* __restrict__ out, int totalHalves) {
  int idx = blockIdx.x * blockDim.x + threadIdx.x;
  if (idx >= totalHalves) return;
  int ntiles = Ndim >> 4;
  int tile = idx >> 9;           // /512 halves per tile
  int within = idx & 511;
  int lane = within >> 4;
  int h = within & 15;
  int kt = tile / ntiles, nt = tile % ntiles;
  int nl = lane & 15, khi = lane >> 4;
  int k = kt * 32 + (h < 8 ? khi * 8 + h : 16 + khi * 8 + (h - 8));
  int n = nt * 16 + nl;
  float v = (k < Kdim) ? W[(long)k * Ndim + n] : 0.0f;
  out[idx] = (_Float16)v;
}

// ---- load A operand: two aligned 16B chunks -> v16h in WMMA A layout ----
__device__ __forceinline__ v16h load_a(const _Float16* p) {
  v8h lo = *(const v8h*)p;
  v8h hi = *(const v8h*)(p + 16);
  return __builtin_shufflevector(lo, hi, 0,1,2,3,4,5,6,7,8,9,10,11,12,13,14,15);
}

// ---- LayerNorm + ReLU over a 16 x (NT*16) D-layout accumulator block ----
// If hrow != nullptr: write post-activation f16 row-major to LDS (stride NT*16).
// Else: return post values in post[nt][j] registers.
template<int NT>
__device__ __forceinline__ void ln_relu(const v8f* acc,
                                        const float* __restrict__ bi,
                                        const float* __restrict__ ga,
                                        const float* __restrict__ be,
                                        int ln, int kh,
                                        _Float16* hrow, float (*post)[8]) {
  constexpr float invH = 1.0f / (float)(NT * 16);
  float bv[NT], gv[NT], ev[NT];
#pragma unroll
  for (int nt = 0; nt < NT; nt++) {
    bv[nt] = bi[nt*16 + ln]; gv[nt] = ga[nt*16 + ln]; ev[nt] = be[nt*16 + ln];
  }
#pragma unroll
  for (int j = 0; j < 8; j++) {
    float v[NT]; float s = 0.f, s2 = 0.f;
#pragma unroll
    for (int nt = 0; nt < NT; nt++) {
      v[nt] = acc[nt][j] + bv[nt];
      s += v[nt]; s2 += v[nt] * v[nt];
    }
#pragma unroll
    for (int m = 1; m < 16; m <<= 1) {     // reduce within each 16-lane half
      s  += __shfl_xor(s,  m, 32);
      s2 += __shfl_xor(s2, m, 32);
    }
    float mu = s * invH;
    float r  = rsqrtf(fmaxf(s2 * invH - mu * mu, 0.f) + 1e-5f);
#pragma unroll
    for (int nt = 0; nt < NT; nt++) {
      float o = fmaxf((v[nt] - mu) * r * gv[nt] + ev[nt], 0.f);
      if (hrow) hrow[(j + 8*kh) * (NT*16) + nt*16 + ln] = (_Float16)o;
      else      post[nt][j] = o;
    }
  }
}

// ---------------- main fused MLP-VAD kernel ----------------
__global__ __launch_bounds__(256) void MLPVAD_main(
    const float* __restrict__ x, const _Float16* __restrict__ wtiles,
    const float* __restrict__ b1, const float* __restrict__ g1, const float* __restrict__ e1,
    const float* __restrict__ b2, const float* __restrict__ g2, const float* __restrict__ e2,
    const float* __restrict__ b3, const float* __restrict__ g3, const float* __restrict__ e3,
    const float* __restrict__ W4, const float* __restrict__ b4,
    float* __restrict__ out, int tilesPerWave)
{
  extern __shared__ _Float16 smem[];
  _Float16* wlds = smem;
  const int tid  = threadIdx.x;
  const int wave = tid >> 5;
  const int l    = tid & 31;
  const int ln   = l & 15;        // N lane within half
  const int kh   = l >> 4;        // K-half / M-half selector
  _Float16* stage = smem + W_LDS_HALVES + wave * STAGE_HALVES;
  _Float16* hbuf  = smem + W_LDS_HALVES + WAVES * STAGE_HALVES + wave * HBUF_HALVES;

  // ---- one-shot weights -> LDS (TDM tensor DMA; fallback: vector copy) ----
#if USE_TDM
  if (tid < 32) {
    // D# group0: count=1, lds_addr, 57-bit global addr, type=2 (bits 127:126)
    unsigned long long ga = (unsigned long long)(uintptr_t)wtiles;
    unsigned la = (unsigned)(uintptr_t)wlds;
    u32x4 g0; i32x8 gA; i32x4 gB = {0,0,0,0}, gC = {0,0,0,0};
    i32x8 gD = {0,0,0,0,0,0,0,0};
    g0[0] = 1u;
    g0[1] = la;
    g0[2] = (unsigned)ga;
    g0[3] = ((unsigned)(ga >> 32) & 0x1FFFFFFu) | 0x80000000u;
    // group1: data_size=4B (code 2); 2D tensor 4096 x 10 dwords, tile = whole
    gA[0] = 0x20000;                 // wg_mask=0, data_size=2
    gA[1] = (int)(4096u << 16);      // tensor_dim0[15:0] into bits 63:48
    gA[2] = (int)(10u << 16);        // tensor_dim1[15:0] into bits 95:80
    gA[3] = (int)(4096u << 16);      // tile_dim0 = 4096
    gA[4] = 10;                      // tile_dim1 = 10, tile_dim2 = 0
    gA[5] = 4096;                    // tensor_dim0_stride = 4096 dwords
    gA[6] = 0; gA[7] = 0;
    __builtin_amdgcn_tensor_load_to_lds(g0, gA, gB, gC, gD, 0);
    __builtin_amdgcn_s_wait_tensorcnt(0);
  }
#else
  {
    const float4* src = (const float4*)wtiles;
    float4* dst = (float4*)wlds;
    for (int i = tid; i < (W_LDS_HALVES * 2 / 16); i += 256) dst[i] = src[i];
  }
#endif
  __syncthreads();

  const int waveGlobal = blockIdx.x * WAVES + wave;
  for (int it = 0; it < tilesPerWave; ++it) {
    const int tile = waveGlobal * tilesPerWave + it;
    const int b  = tile >> 9;               // T/16 = 512 tiles per batch row
    const int t0 = (tile & 511) << 4;

    // ---- stage 26 context frames (t0-5 .. t0+20) as f16; zero-pad edges ----
    for (int i = l; i < STAGE_HALVES; i += 32) {
      float v = 0.0f;
      if (i < 26 * NMELS) {
        int f = t0 - 5 + i / NMELS;
        int m = i % NMELS;
        if (f >= 0 && f < T_LEN) v = x[((long)b * T_LEN + f) * NMELS + m];
      }
      stage[i] = (_Float16)v;
    }
    __syncthreads();

    // ---- Layer 1: (16 x 448) @ (448 x 128), K padded with zero weights ----
    v8f acc1[NT1] = {};
#pragma unroll
    for (int kt = 0; kt < KT1; kt++) {
      v16h a = load_a(stage + ln * NMELS + kt * 32 + kh * 8);
#pragma unroll
      for (int nt = 0; nt < NT1; nt++) {
        v16h bm = *(const v16h*)(wlds + (kt * NT1 + nt) * 512 + l * 16);
        acc1[nt] = __builtin_amdgcn_wmma_f32_16x16x32_f16(
            false, a, false, bm, (short)0, acc1[nt], false, false);
      }
    }
    ln_relu<NT1>(acc1, b1, g1, e1, ln, kh, hbuf, (float(*)[8])nullptr);
    __syncthreads();

    // ---- Layer 2: (16 x 128) @ (128 x 128) ----
    v8f acc2[NT2] = {};
#pragma unroll
    for (int kt = 0; kt < KT2; kt++) {
      v16h a = load_a(hbuf + ln * 128 + kt * 32 + kh * 8);
#pragma unroll
      for (int nt = 0; nt < NT2; nt++) {
        v16h bm = *(const v16h*)(wlds + W1_HALVES + (kt * NT2 + nt) * 512 + l * 16);
        acc2[nt] = __builtin_amdgcn_wmma_f32_16x16x32_f16(
            false, a, false, bm, (short)0, acc2[nt], false, false);
      }
    }
    ln_relu<NT2>(acc2, b2, g2, e2, ln, kh, hbuf, (float(*)[8])nullptr);
    __syncthreads();

    // ---- Layer 3: (16 x 128) @ (128 x 64) ----
    v8f acc3[NT3] = {};
#pragma unroll
    for (int kt = 0; kt < KT3; kt++) {
      v16h a = load_a(hbuf + ln * 128 + kt * 32 + kh * 8);
#pragma unroll
      for (int nt = 0; nt < NT3; nt++) {
        v16h bm = *(const v16h*)(wlds + W1_HALVES + W2_HALVES + (kt * NT3 + nt) * 512 + l * 16);
        acc3[nt] = __builtin_amdgcn_wmma_f32_16x16x32_f16(
            false, a, false, bm, (short)0, acc3[nt], false, false);
      }
    }
    float post[NT3][8];
    ln_relu<NT3>(acc3, b3, g3, e3, ln, kh, (_Float16*)nullptr, post);

    // ---- Layer 4: (16 x 64) @ (64 x 1) + b4, register dot + shuffle reduce ----
    float w4v[NT3];
#pragma unroll
    for (int nt = 0; nt < NT3; nt++) w4v[nt] = W4[nt * 16 + ln];
    const float bias4 = b4[0];
#pragma unroll
    for (int j = 0; j < 8; j++) {
      float p = 0.f;
#pragma unroll
      for (int nt = 0; nt < NT3; nt++) p += post[nt][j] * w4v[nt];
#pragma unroll
      for (int m = 1; m < 16; m <<= 1) p += __shfl_xor(p, m, 32);
      if (ln == 0) out[tile * 16 + j + 8 * kh] = p + bias4;
    }
    __syncthreads();
  }
}

extern "C" void kernel_launch(void* const* d_in, const int* in_sizes, int n_in,
                              void* d_out, int out_size, void* d_ws, size_t ws_size,
                              hipStream_t stream) {
  const float* x  = (const float*)d_in[0];
  const float* W1 = (const float*)d_in[1];
  const float* b1 = (const float*)d_in[2];
  const float* g1 = (const float*)d_in[3];
  const float* e1 = (const float*)d_in[4];
  const float* W2 = (const float*)d_in[5];
  const float* b2 = (const float*)d_in[6];
  const float* g2 = (const float*)d_in[7];
  const float* e2 = (const float*)d_in[8];
  const float* W3 = (const float*)d_in[9];
  const float* b3 = (const float*)d_in[10];
  const float* g3 = (const float*)d_in[11];
  const float* e3 = (const float*)d_in[12];
  const float* W4 = (const float*)d_in[13];
  const float* b4 = (const float*)d_in[14];

  _Float16* wt = (_Float16*)d_ws;   // 163840 B of f16 weight tiles

  MLPVAD_prep_wtiles<<<W1_HALVES / 256, 256, 0, stream>>>(W1, 440, 128, wt, W1_HALVES);
  MLPVAD_prep_wtiles<<<W2_HALVES / 256, 256, 0, stream>>>(W2, 128, 128, wt + W1_HALVES, W2_HALVES);
  MLPVAD_prep_wtiles<<<W3_HALVES / 256, 256, 0, stream>>>(W3, 128,  64, wt + W1_HALVES + W2_HALVES, W3_HALVES);

  (void)hipFuncSetAttribute(reinterpret_cast<const void*>(MLPVAD_main),
                            hipFuncAttributeMaxDynamicSharedMemorySize, SMEM_BYTES);

  // 16384 row-tiles total; 512 blocks x 8 waves x 4 tiles/wave
  MLPVAD_main<<<512, 256, SMEM_BYTES, stream>>>(
      x, wt, b1, g1, e1, b2, g2, e2, b3, g3, e3, W4, b4,
      (float*)d_out, 4);
}
